// Encoder_3126736191824
// MI455X (gfx1250) — matrix operand order repaired and checked by reading, not compiled, not run on hardware
//
#include <hip/hip_runtime.h>

#define DMODEL 1024
#define NHEAD  16
#define HDIM   64
#define FFDIM  4096
#define SEQ    2048
#define NBATCH 2
#define NTOK   (NBATCH * SEQ)

typedef __attribute__((ext_vector_type(16))) __bf16 v16bf;
typedef __attribute__((ext_vector_type(8)))  float  v8f;
typedef __attribute__((ext_vector_type(4)))  unsigned int u32x4;
typedef __attribute__((ext_vector_type(8)))  int i32x8;
typedef __attribute__((ext_vector_type(4)))  int i32x4;

union Frag { v16bf bf; unsigned int u[8]; };

__device__ __forceinline__ unsigned short f2bf(float f) {
  unsigned int u = __float_as_uint(f);
  u += 0x7FFFu + ((u >> 16) & 1u);          // round-to-nearest-even
  return (unsigned short)(u >> 16);
}

// A-matrix (16x32 bf16) k index for VGPR v, lane-half h (ISA 7.12.2)
__device__ __forceinline__ int kbA(int v, int h) {
  return (v < 4 ? 2 * v : 16 + 2 * (v - 4)) + 8 * h;
}
// B-matrix (32x16 bf16) k index for VGPR v, lane-half h
__device__ __forceinline__ int kbB(int v, int h) {
  return (v < 4 ? 2 * v : 16 + 2 * (v - 4)) + 8 * h;
}

// TDM: DMA a 2D bf16 tile [tile_rows x 32] (row stride = K elements) into LDS.
// D# layout per CDNA5 ISA.
__device__ __forceinline__ void tdm_load_2d(unsigned int lds_off,
                                            const void* gptr,
                                            int tile_rows, int K_elems) {
  unsigned long long ga = (unsigned long long)(size_t)gptr;
  u32x4 g0;
  g0[0] = 1u;                                     // count=1, user mode
  g0[1] = lds_off;                                // lds_addr [63:32]
  g0[2] = (unsigned int)ga;                       // global_addr low
  g0[3] = (unsigned int)(ga >> 32) | (2u << 30);  // addr hi | type=2 (image)
  i32x8 g1;
  g1[0] = 1 << 16;                  // data_size=1 (2 bytes/elem)
  g1[1] = 32 << 16;                 // tensor_dim0[15:0] = 32  (bits 63:48)
  g1[2] = tile_rows << 16;          // tensor_dim0 hi=0 | tensor_dim1 lo
  g1[3] = 32 << 16;                 // tensor_dim1 hi=0 | tile_dim0 = 32
  g1[4] = tile_rows & 0xFFFF;       // tile_dim1 | tile_dim2=0
  g1[5] = K_elems;                  // tensor_dim0_stride low32
  g1[6] = 0;                        // stride hi | tensor_dim1_stride lo
  g1[7] = 0;
  i32x4 z4 = {0, 0, 0, 0};
#if __clang_major__ >= 23
  i32x8 z8 = {0, 0, 0, 0, 0, 0, 0, 0};
  __builtin_amdgcn_tensor_load_to_lds(g0, g1, z4, z4, z8, 0);
#else
  __builtin_amdgcn_tensor_load_to_lds(g0, g1, z4, z4, 0);
#endif
}

// ---------------------------------------------------------------- converters
__global__ void cvt_f32_bf16(const float* __restrict__ in,
                             unsigned short* __restrict__ out, int n) {
  int i = blockIdx.x * 256 + threadIdx.x;
  if (i < n) out[i] = f2bf(in[i]);
}

// w [K][N] fp32 -> wt [N][K] bf16
__global__ void transpose_cvt(const float* __restrict__ w,
                              unsigned short* __restrict__ wt, int K, int N) {
  long long i = (long long)blockIdx.x * 256 + threadIdx.x;
  if (i >= (long long)K * N) return;
  int n = (int)(i / K), k = (int)(i % K);
  wt[i] = f2bf(w[(long long)k * N + n]);
}

// ---------------------------------------------------------------- WMMA GEMM
// C[M,N] = A[M,K](bf16) * Bt[N,K](bf16)^T (+bias)(+relu)(+resid f32)
// block: 256 thr = 8 waves, tile 128x64; wave tile 32x32 (2x2 WMMA).
// Tiles staged by the Tensor Data Mover, double-buffered in LDS.
__global__ __launch_bounds__(256)
void gemm_bf16(const unsigned short* __restrict__ A,
               const unsigned short* __restrict__ Bt,
               const float* __restrict__ bias,
               const float* __restrict__ resid,
               unsigned short* __restrict__ outb,
               float* __restrict__ outf,
               int M, int N, int K, int relu) {
  __shared__ __align__(16) unsigned short lds_a[2][128][32];
  __shared__ __align__(16) unsigned short lds_b[2][64][32];

  const int tid  = threadIdx.x;
  const int lane = tid & 31;
  const int wid  = tid >> 5;
  const int wm   = wid & 3;        // wave row 0..3  (32 rows each)
  const int wn   = wid >> 2;       // wave col 0..1  (32 cols each)
  const int half = lane >> 4;
  const int ln15 = lane & 15;
  const int rowBase = blockIdx.y * 128;
  const int colBase = blockIdx.x * 64;

  const unsigned int aoff[2] = {(unsigned int)(size_t)&lds_a[0][0][0],
                                (unsigned int)(size_t)&lds_a[1][0][0]};
  const unsigned int boff[2] = {(unsigned int)(size_t)&lds_b[0][0][0],
                                (unsigned int)(size_t)&lds_b[1][0][0]};
  const unsigned short* abase = A + (size_t)rowBase * K;
  const unsigned short* bbase = Bt + (size_t)colBase * K;

  v8f zero8 = {0.f, 0.f, 0.f, 0.f, 0.f, 0.f, 0.f, 0.f};
  v8f c[2][2];
  c[0][0] = zero8; c[0][1] = zero8; c[1][0] = zero8; c[1][1] = zero8;

  const int steps = K / 32;
  if (wid == 0) {
    tdm_load_2d(aoff[0], abase, 128, K);
    tdm_load_2d(boff[0], bbase, 64, K);
  }

  for (int i = 0; i < steps; ++i) {
    if (wid == 0) __builtin_amdgcn_s_wait_tensorcnt(0);
    __syncthreads();   // stage i visible; all waves done reading buf (i&1)^1
    const int cur = i & 1;
    if (wid == 0 && i + 1 < steps) {
      tdm_load_2d(aoff[cur ^ 1], abase + (i + 1) * 32, 128, K);
      tdm_load_2d(boff[cur ^ 1], bbase + (i + 1) * 32, 64, K);
    }

    Frag a[2], b[2];
#pragma unroll
    for (int mi = 0; mi < 2; ++mi)
#pragma unroll
      for (int v = 0; v < 8; ++v)
        a[mi].u[v] =
            *(const unsigned int*)&lds_a[cur][wm * 32 + mi * 16 + ln15][kbA(v, half)];
#pragma unroll
    for (int ni = 0; ni < 2; ++ni)
#pragma unroll
      for (int v = 0; v < 8; ++v)
        b[ni].u[v] =
            *(const unsigned int*)&lds_b[cur][wn * 32 + ni * 16 + ln15][kbB(v, half)];

#pragma unroll
    for (int mi = 0; mi < 2; ++mi)
#pragma unroll
      for (int ni = 0; ni < 2; ++ni)
        c[mi][ni] = __builtin_amdgcn_wmma_f32_16x16x32_bf16(
            false, a[mi].bf, false, b[ni].bf, (short)0, c[mi][ni], false, false);
  }

#pragma unroll
  for (int mi = 0; mi < 2; ++mi)
#pragma unroll
    for (int ni = 0; ni < 2; ++ni) {
      const int col = colBase + wn * 32 + ni * 16 + ln15;
      const float bv = bias ? bias[col] : 0.f;
#pragma unroll
      for (int r = 0; r < 8; ++r) {
        const int row = rowBase + wm * 32 + mi * 16 + 8 * half + r;
        float v = c[mi][ni][r] + bv;
        if (relu) v = fmaxf(v, 0.f);
        size_t idx = (size_t)row * N + col;
        if (resid) v += resid[idx];
        if (outf) outf[idx] = v;
        if (outb) outb[idx] = f2bf(v);
      }
    }
}

// ---------------------------------------------------------------- attention
// flash-style: block = 256 thr (8 waves), 128 queries/block, 16 per wave.
// qkv bf16 [NTOK][3*DMODEL]; O bf16 [NTOK][DMODEL]
__global__ __launch_bounds__(256)
void flash_attn(const unsigned short* __restrict__ qkv,
                unsigned short* __restrict__ O) {
  __shared__ __align__(16) unsigned short k_tile[64][64];   // [key][d]
  __shared__ __align__(16) unsigned short vt_tile[64][64];  // [d][key]
  __shared__ __align__(16) unsigned short p_tile[8][16][64];

  const int tid  = threadIdx.x;
  const int lane = tid & 31;
  const int wid  = tid >> 5;
  const int half = lane >> 4;
  const int ln15 = lane & 15;
  const int b = blockIdx.z, h = blockIdx.y;
  const int qbase = blockIdx.x * 128 + wid * 16;
  const size_t rstride = 3 * DMODEL;

  // Q fragments: row = ln15, contraction over d (two 32-chunks)
  const unsigned short* qp =
      qkv + (size_t)(b * SEQ + qbase + ln15) * rstride + h * HDIM;
  Frag q[2];
#pragma unroll
  for (int kc = 0; kc < 2; ++kc)
#pragma unroll
    for (int v = 0; v < 8; ++v)
      q[kc].u[v] = *(const unsigned int*)(qp + kc * 32 + kbA(v, half));

  float run_m[8], run_l[8];
  v8f zero8 = {0.f, 0.f, 0.f, 0.f, 0.f, 0.f, 0.f, 0.f};
  v8f o[4];
#pragma unroll
  for (int r = 0; r < 8; ++r) { run_m[r] = -1e30f; run_l[r] = 0.f; }
#pragma unroll
  for (int nd = 0; nd < 4; ++nd) o[nd] = zero8;

  const int lkk = tid >> 2;           // K loader: key row 0..63
  const int ld  = (tid & 3) * 16;     // K loader: d segment
  const int vd  = tid & 63;           // V loader: d
  const int vk0 = (tid >> 6) * 16;    // V loader: key base

  for (int ch = 0; ch < SEQ / 64; ++ch) {
    const int kb0 = ch * 64;
    const uint4* kg = (const uint4*)(qkv +
        (size_t)(b * SEQ + kb0 + lkk) * rstride + DMODEL + h * HDIM + ld);
    uint4 kv0 = kg[0], kv1 = kg[1];
    unsigned short vt[16];
    const unsigned short* vg = qkv +
        (size_t)(b * SEQ + kb0 + vk0) * rstride + 2 * DMODEL + h * HDIM + vd;
#pragma unroll
    for (int i = 0; i < 16; ++i) vt[i] = vg[(size_t)i * rstride];

    __syncthreads();
    *(uint4*)&k_tile[lkk][ld]     = kv0;
    *(uint4*)&k_tile[lkk][ld + 8] = kv1;
    unsigned int pw[8];
#pragma unroll
    for (int j = 0; j < 8; ++j)
      pw[j] = (unsigned int)vt[2 * j] | ((unsigned int)vt[2 * j + 1] << 16);
    *(uint4*)&vt_tile[vd][vk0]     = make_uint4(pw[0], pw[1], pw[2], pw[3]);
    *(uint4*)&vt_tile[vd][vk0 + 8] = make_uint4(pw[4], pw[5], pw[6], pw[7]);
    __syncthreads();

    // S = Q*K^T / sqrt(HD)
    v8f s[4];
#pragma unroll
    for (int ni = 0; ni < 4; ++ni) {
      Frag kf0, kf1;
#pragma unroll
      for (int v = 0; v < 8; ++v) {
        int kb = kbB(v, half);
        kf0.u[v] = *(const unsigned int*)&k_tile[ni * 16 + ln15][kb];
        kf1.u[v] = *(const unsigned int*)&k_tile[ni * 16 + ln15][32 + kb];
      }
      s[ni] = __builtin_amdgcn_wmma_f32_16x16x32_bf16(
          false, q[0].bf, false, kf0.bf, (short)0, zero8, false, false);
      s[ni] = __builtin_amdgcn_wmma_f32_16x16x32_bf16(
          false, q[1].bf, false, kf1.bf, (short)0, s[ni], false, false);
      s[ni] *= 0.125f;   // 1/sqrt(64)
    }

    // online softmax over this 64-key chunk (rows live in C layout)
    float cm[8];
#pragma unroll
    for (int r = 0; r < 8; ++r)
      cm[r] = fmaxf(fmaxf(s[0][r], s[1][r]), fmaxf(s[2][r], s[3][r]));
    for (int off = 1; off < 16; off <<= 1)
#pragma unroll
      for (int r = 0; r < 8; ++r)
        cm[r] = fmaxf(cm[r], __shfl_xor(cm[r], off, 32));

    float rs[8];
#pragma unroll
    for (int r = 0; r < 8; ++r) {
      float nm = fmaxf(run_m[r], cm[r]);
      float al = __expf(run_m[r] - nm);
      run_m[r] = nm;
      run_l[r] *= al;
#pragma unroll
      for (int nd = 0; nd < 4; ++nd) o[nd][r] *= al;
      float sum = 0.f;
#pragma unroll
      for (int ni = 0; ni < 4; ++ni) {
        float pv = __expf(s[ni][r] - nm);
        sum += pv;
        p_tile[wid][8 * half + r][ni * 16 + ln15] = f2bf(pv);
      }
      rs[r] = sum;
    }
    for (int off = 1; off < 16; off <<= 1)
#pragma unroll
      for (int r = 0; r < 8; ++r) rs[r] += __shfl_xor(rs[r], off, 32);
#pragma unroll
    for (int r = 0; r < 8; ++r) run_l[r] += rs[r];

    // O += P*V  (contraction over 64 keys, per-wave p_tile: no x-wave hazard)
    Frag pa[2];
#pragma unroll
    for (int kc = 0; kc < 2; ++kc)
#pragma unroll
      for (int v = 0; v < 8; ++v)
        pa[kc].u[v] =
            *(const unsigned int*)&p_tile[wid][ln15][kc * 32 + kbA(v, half)];
#pragma unroll
    for (int nd = 0; nd < 4; ++nd) {
      Frag vb0, vb1;
#pragma unroll
      for (int v = 0; v < 8; ++v) {
        int kb = kbB(v, half);
        vb0.u[v] = *(const unsigned int*)&vt_tile[nd * 16 + ln15][kb];
        vb1.u[v] = *(const unsigned int*)&vt_tile[nd * 16 + ln15][32 + kb];
      }
      o[nd] = __builtin_amdgcn_wmma_f32_16x16x32_bf16(
          false, pa[0].bf, false, vb0.bf, (short)0, o[nd], false, false);
      o[nd] = __builtin_amdgcn_wmma_f32_16x16x32_bf16(
          false, pa[1].bf, false, vb1.bf, (short)0, o[nd], false, false);
    }
  }

#pragma unroll
  for (int nd = 0; nd < 4; ++nd)
#pragma unroll
    for (int r = 0; r < 8; ++r) {
      int tok = b * SEQ + qbase + 8 * half + r;
      int col = h * HDIM + nd * 16 + ln15;
      O[(size_t)tok * DMODEL + col] = f2bf(o[nd][r] / run_l[r]);
    }
}

// ---------------------------------------------------------------- layernorm
__global__ __launch_bounds__(256)
void layernorm_k(const float* __restrict__ in, const float* __restrict__ gamma,
                 const float* __restrict__ beta, float* __restrict__ outf,
                 unsigned short* __restrict__ outb) {
  __shared__ float red[256];
  const int row = blockIdx.x, tid = threadIdx.x;
  const float* x = in + (size_t)row * DMODEL;
  float v[4], s = 0.f;
#pragma unroll
  for (int i = 0; i < 4; ++i) { v[i] = x[tid + i * 256]; s += v[i]; }
  red[tid] = s; __syncthreads();
  for (int st = 128; st > 0; st >>= 1) {
    if (tid < st) red[tid] += red[tid + st];
    __syncthreads();
  }
  float mean = red[0] * (1.f / DMODEL);
  __syncthreads();
  float sq = 0.f;
#pragma unroll
  for (int i = 0; i < 4; ++i) { float d = v[i] - mean; sq += d * d; }
  red[tid] = sq; __syncthreads();
  for (int st = 128; st > 0; st >>= 1) {
    if (tid < st) red[tid] += red[tid + st];
    __syncthreads();
  }
  float rstd = rsqrtf(red[0] * (1.f / DMODEL) + 1e-5f);
#pragma unroll
  for (int i = 0; i < 4; ++i) {
    int c = tid + i * 256;
    float y = (v[i] - mean) * rstd * gamma[c] + beta[c];
    size_t idx = (size_t)row * DMODEL + c;
    if (outf) outf[idx] = y;
    if (outb) outb[idx] = f2bf(y);
  }
}

// ---------------------------------------------------------------- launcher
extern "C" void kernel_launch(void* const* d_in, const int* in_sizes, int n_in,
                              void* d_out, int out_size, void* d_ws, size_t ws_size,
                              hipStream_t stream) {
  const float* x     = (const float*)d_in[0];
  const float* w_qkv = (const float*)d_in[1];
  const float* b_qkv = (const float*)d_in[2];
  const float* w_o   = (const float*)d_in[3];
  const float* b_o   = (const float*)d_in[4];
  const float* g1    = (const float*)d_in[5];
  const float* be1   = (const float*)d_in[6];
  const float* w1    = (const float*)d_in[7];
  const float* b1    = (const float*)d_in[8];
  const float* w2    = (const float*)d_in[9];
  const float* b2    = (const float*)d_in[10];
  const float* g2    = (const float*)d_in[11];
  const float* be2   = (const float*)d_in[12];
  float* out = (float*)d_out;

  char* ws = (char*)d_ws;
  size_t off = 0;
  auto alloc = [&](size_t bytes) {
    char* p = ws + off;
    off += (bytes + 255) & ~(size_t)255;
    return p;
  };
  unsigned short* xb    = (unsigned short*)alloc((size_t)NTOK * DMODEL * 2);
  unsigned short* wqkvT = (unsigned short*)alloc((size_t)3 * DMODEL * DMODEL * 2);
  unsigned short* woT   = (unsigned short*)alloc((size_t)DMODEL * DMODEL * 2);
  unsigned short* w1T   = (unsigned short*)alloc((size_t)FFDIM * DMODEL * 2);
  unsigned short* w2T   = (unsigned short*)alloc((size_t)DMODEL * FFDIM * 2);
  unsigned short* qkvb  = (unsigned short*)alloc((size_t)NTOK * 3 * DMODEL * 2);
  unsigned short* Ob    = (unsigned short*)alloc((size_t)NTOK * DMODEL * 2);
  float*          hf    = (float*)alloc((size_t)NTOK * DMODEL * 4);
  unsigned short* hb    = (unsigned short*)alloc((size_t)NTOK * DMODEL * 2);
  unsigned short* ffb   = (unsigned short*)alloc((size_t)NTOK * FFDIM * 2);
  float*          buf   = (float*)alloc((size_t)NTOK * DMODEL * 4);
  (void)ws_size; (void)in_sizes; (void)n_in; (void)out_size;

  const int nx = NTOK * DMODEL;
  cvt_f32_bf16<<<(nx + 255) / 256, 256, 0, stream>>>(x, xb, nx);
  transpose_cvt<<<(3 * DMODEL * DMODEL + 255) / 256, 256, 0, stream>>>(
      w_qkv, wqkvT, DMODEL, 3 * DMODEL);
  transpose_cvt<<<(DMODEL * DMODEL + 255) / 256, 256, 0, stream>>>(
      w_o, woT, DMODEL, DMODEL);
  transpose_cvt<<<(DMODEL * FFDIM + 255) / 256, 256, 0, stream>>>(
      w1, w1T, DMODEL, FFDIM);
  transpose_cvt<<<(FFDIM * DMODEL + 255) / 256, 256, 0, stream>>>(
      w2, w2T, FFDIM, DMODEL);

  // qkv = x @ w_qkv + b_qkv              -> bf16 [NTOK, 3D]
  gemm_bf16<<<dim3(3 * DMODEL / 64, NTOK / 128), 256, 0, stream>>>(
      xb, wqkvT, b_qkv, nullptr, qkvb, nullptr, NTOK, 3 * DMODEL, DMODEL, 0);

  // attention -> Ob bf16 [NTOK, D]
  flash_attn<<<dim3(SEQ / 128, NHEAD, NBATCH), 256, 0, stream>>>(qkvb, Ob);

  // proj = O @ w_o + b_o + x             -> f32 buf
  gemm_bf16<<<dim3(DMODEL / 64, NTOK / 128), 256, 0, stream>>>(
      Ob, woT, b_o, x, nullptr, buf, NTOK, DMODEL, DMODEL, 0);

  // h = LN1(buf)                          -> hf f32, hb bf16
  layernorm_k<<<NTOK, 256, 0, stream>>>(buf, g1, be1, hf, hb);

  // ff = relu(h @ w1 + b1)                -> ffb bf16
  gemm_bf16<<<dim3(FFDIM / 64, NTOK / 128), 256, 0, stream>>>(
      hb, w1T, b1, nullptr, ffb, nullptr, NTOK, FFDIM, DMODEL, 1);

  // y = ff @ w2 + b2 + h                  -> f32 buf
  gemm_bf16<<<dim3(DMODEL / 64, NTOK / 128), 256, 0, stream>>>(
      ffb, w2T, b2, hf, nullptr, buf, NTOK, DMODEL, FFDIM, 0);

  // out = LN2(buf)
  layernorm_k<<<NTOK, 256, 0, stream>>>(buf, g2, be2, out, nullptr);
}
